// STKDiff_85048942395468
// MI455X (gfx1250) — matrix operand order repaired and checked by reading, not compiled, not run on hardware
//
#include <hip/hip_runtime.h>
#include <math.h>

// ---------------------------------------------------------------------------
// Problem constants (from the reference)
// ---------------------------------------------------------------------------
#define Bn   8
#define Cn   128
#define Kn   256
#define Ln   168
#define En   85          // rfft bins
#define Ep   96          // E padded to 6x16 tiles
#define FSTR 192         // padded complex row: re in [0,96), im in [96,192)
#define RBCK (Bn*Cn*Kn)  // 262144 rows (b,c,k)
#define RKE  (Bn*Kn*Ep)  // 196608 rows (b,k,e) padded

typedef __attribute__((ext_vector_type(16))) _Float16 v16h;
typedef __attribute__((ext_vector_type(8)))  float    v8f;

// ---------------------------------------------------------------------------
// WMMA helper: 16x16x32 f16 -> f32, wave32.  Per ISA 7.12, f16 WMMA has no
// A/B negation modifier (NEG = {CNeg,0,0}); subtraction is done by negating
// the fragment with packed VALU ops (negh) before the wmma.
// ---------------------------------------------------------------------------
static __device__ __forceinline__ v8f wmma16(v16h a, v16h b, v8f c) {
  return __builtin_amdgcn_wmma_f32_16x16x32_f16(false, a, false, b, (short)0, c,
                                                false, false);
}

static __device__ __forceinline__ v16h negh(v16h a) {
  v16h r;
#pragma unroll
  for (int i = 0; i < 16; ++i) r[i] = -a[i];
  return r;
}

// A fragment 16x32 (MxK), row-major source.  Per ISA 7.12.2:
// lanes 0-15 -> M=lane, half=lane>>4 selects K groups {0..7,16..23}/{8..15,24..31}.
static __device__ __forceinline__ v16h load_a(const _Float16* src, int stride) {
  int lane = threadIdx.x & 31;
  int m = lane & 15, half = lane >> 4;
  v16h a;
#pragma unroll
  for (int v = 0; v < 8; ++v) {
    int k = (v < 4 ? 2 * v : 16 + 2 * (v - 4)) + 8 * half;
    a[2 * v]     = src[(long)m * stride + k];
    a[2 * v + 1] = src[(long)m * stride + k + 1];
  }
  return a;
}

// B fragment 32x16 (KxN), row-major source: lane holds column N=lane&15, K in table order.
static __device__ __forceinline__ v16h load_b(const _Float16* src, int stride) {
  int lane = threadIdx.x & 31;
  int n = lane & 15, half = lane >> 4;
  v16h b;
#pragma unroll
  for (int j = 0; j < 16; ++j) { int k = (j < 8 ? j : j + 8) + 8 * half; b[j] = src[(long)k * stride + n]; }
  return b;
}

// B fragment from a transposed source: B[k][n] = src[n*stride + k].
static __device__ __forceinline__ v16h load_bt(const _Float16* src, int stride) {
  int lane = threadIdx.x & 31;
  int n = lane & 15, half = lane >> 4;
  v16h b;
#pragma unroll
  for (int j = 0; j < 16; ++j) { int k = (j < 8 ? j : j + 8) + 8 * half; b[j] = src[(long)n * stride + k]; }
  return b;
}

// ---------------------------------------------------------------------------
// Prep: f16 DFT twiddle matrices + transposed f16 weight copies.
// Tfwd [192][192]: col e<85 = cos(2*pi*e*l/L); col 96+e = -sin(...); rows l>=168 zero.
// Tinv [192][176]: row e<85 = w_e*cos/L; row 96+e = -w_e*sin/L; cols l>=168 zero.
// ---------------------------------------------------------------------------
__global__ void k_prep(_Float16* Tfwd, _Float16* Tinv,
                       const float* Wq_re, const float* Wq_im,
                       const float* Wk_re, const float* Wk_im,
                       const float* Wv_re, const float* Wv_im,
                       const float* fw1, const float* fw2,
                       _Float16* WQtre, _Float16* WQtim,
                       _Float16* WKtre, _Float16* WKtim,
                       _Float16* WVtre, _Float16* WVtim,
                       _Float16* FW1t, _Float16* FW2t) {
  const float PI2 = 6.28318530717958647692f;
  int tot = 192 * 192 + 192 * 176 + 6 * 256 * 128 + 2 * 128 * 128;
  for (int idx = blockIdx.x * blockDim.x + threadIdx.x; idx < tot;
       idx += gridDim.x * blockDim.x) {
    int i = idx;
    if (i < 192 * 192) {
      int l = i / 192, e = i % 192;
      float v = 0.f;
      if (l < Ln) {
        if (e < En)                      v =  cosf(PI2 * (float)(e * l) / (float)Ln);
        else if (e >= 96 && e < 96 + En) v = -sinf(PI2 * (float)((e - 96) * l) / (float)Ln);
      }
      Tfwd[i] = (_Float16)v;
      continue;
    }
    i -= 192 * 192;
    if (i < 192 * 176) {
      int r = i / 176, l = i % 176;
      float v = 0.f;
      if (l < Ln) {
        if (r < En) {
          float w = (r == 0 || r == En - 1) ? 1.f : 2.f;
          v = w * cosf(PI2 * (float)(r * l) / (float)Ln) / (float)Ln;
        } else if (r >= 96 && r < 96 + En) {
          int e = r - 96;
          float w = (e == 0 || e == En - 1) ? 1.f : 2.f;
          v = -w * sinf(PI2 * (float)(e * l) / (float)Ln) / (float)Ln;
        }
      }
      Tinv[i] = (_Float16)v;
      continue;
    }
    i -= 192 * 176;
    if (i < 6 * 256 * 128) {   // Wt[ci][co] = W[co][ci], W is [128][256] f32
      int w = i / (256 * 128), j = i % (256 * 128);
      int ci = j / 128, co = j % 128;
      const float* W = (w == 0) ? Wq_re : (w == 1) ? Wq_im : (w == 2) ? Wk_re
                      : (w == 3) ? Wk_im : (w == 4) ? Wv_re : Wv_im;
      _Float16* D = (w == 0) ? WQtre : (w == 1) ? WQtim : (w == 2) ? WKtre
                   : (w == 3) ? WKtim : (w == 4) ? WVtre : WVtim;
      D[j] = (_Float16)W[co * 256 + ci];
      continue;
    }
    i -= 6 * 256 * 128;
    {
      int w = i / (128 * 128), j = i % (128 * 128);
      int ci = j / 128, co = j % 128;
      const float* W = w ? fw2 : fw1;
      _Float16* D = w ? FW2t : FW1t;
      D[j] = (_Float16)W[co * 128 + ci];
    }
  }
}

// ---------------------------------------------------------------------------
// DFT-as-GEMM: Out[M][ldo] (f32) = A[M][lda] (f32 or f16, k<Kreal else 0)
//              x Bm[Ktiles*32][ldb] (f16).  One wave per 16x16 tile.
// Used for rfft (both inputs) and the fused irfft at the end.
// ---------------------------------------------------------------------------
template<typename AT>
__global__ void k_dft_gemm(const AT* __restrict__ A, long lda, int Kreal, int Ktiles,
                           const _Float16* __restrict__ Bm, int ldb,
                           float* __restrict__ Out, long ldo, int Nreal) {
  long mo = (long)blockIdx.x * 16;
  int  no = blockIdx.y * 16;
  int lane = threadIdx.x & 31;
  int m = lane & 15, half = lane >> 4;
  v8f acc = {};
  for (int kt = 0; kt < Ktiles; ++kt) {
    int kb = kt * 32;
    v16h a, b;
    const AT* ap = A + (mo + m) * lda;
#pragma unroll
    for (int v = 0; v < 8; ++v) {
      int k = kb + (v < 4 ? 2 * v : 16 + 2 * (v - 4)) + 8 * half;
      a[2 * v]     = (k     < Kreal) ? (_Float16)(float)ap[k]     : (_Float16)0.f;
      a[2 * v + 1] = (k + 1 < Kreal) ? (_Float16)(float)ap[k + 1] : (_Float16)0.f;
    }
    const _Float16* bp = Bm + (long)kb * ldb + no;
#pragma unroll
    for (int j = 0; j < 16; ++j) { int k = (j < 8 ? j : j + 8) + 8 * half; b[j] = bp[(long)k * ldb + m]; }
    acc = wmma16(a, b, acc);
  }
  int n = lane & 15, mb = 8 * half;
  if (no + n < Nreal) {
#pragma unroll
    for (int r = 0; r < 8; ++r)
      Out[(mo + mb + r) * ldo + no + n] = acc[r];
  }
}

// ---------------------------------------------------------------------------
// Top-12 magnitude rank partition per (b,c,k) row.  Iterative top-4 with
// scatter-back == rank buckets: ranks 0-3 -> sub0, 4-7 -> sub1, 8-11 -> sub2,
// rest -> 3 (residual).
// ---------------------------------------------------------------------------
__global__ void k_topk(const float* __restrict__ XFT, unsigned char* __restrict__ BUCK) {
  long row = (long)blockIdx.x * blockDim.x + threadIdx.x;
  if (row >= RBCK) return;
  const float* re = XFT + row * FSTR;
  const float* im = re + 96;
  unsigned char* bk = BUCK + row * Ep;
  for (int e = 0; e < Ep; ++e) bk[e] = 3;
  unsigned int sel[3] = {0u, 0u, 0u};
  for (int r = 0; r < 12; ++r) {
    float best = -1.f; int bi = 0;
    for (int e = 0; e < En; ++e) {
      if ((sel[e >> 5] >> (e & 31)) & 1u) continue;
      float m = re[e] * re[e] + im[e] * im[e];
      if (m > best) { best = m; bi = e; }
    }
    sel[bi >> 5] |= 1u << (bi & 31);
    bk[bi] = (unsigned char)(r >> 2);
  }
}

// ---------------------------------------------------------------------------
// Pack (b,c,k,e) -> (b,k,e,c) f16 operands for the linears + |x_ft| gate input.
// x_ft pad columns (e>=85) are zero by construction of Tfwd.
// ---------------------------------------------------------------------------
__global__ void k_pack_kv(const float* __restrict__ XFT, const float* __restrict__ BSF,
                          _Float16* __restrict__ AXre, _Float16* __restrict__ AXim,
                          _Float16* __restrict__ ABre, _Float16* __restrict__ ABim,
                          _Float16* __restrict__ AG) {
  long idx = (long)blockIdx.x * blockDim.x + threadIdx.x;
  if (idx >= (long)RKE * 128) return;
  int c = (int)(idx & 127);
  long row = idx >> 7;
  int e = (int)(row % Ep);
  long bk = row / Ep;
  int b = (int)(bk / Kn), k = (int)(bk % Kn);
  long src = ((long)(b * Cn + c) * Kn + k) * FSTR;
  float xr = XFT[src + e], xi = XFT[src + 96 + e];
  AXre[idx] = (_Float16)xr;
  AXim[idx] = (_Float16)xi;
  ABre[idx] = (_Float16)BSF[src + e];
  ABim[idx] = (_Float16)BSF[src + 96 + e];
  AG[idx]   = (_Float16)sqrtf(xr * xr + xi * xi);
}

__global__ void k_pack_q(const float* __restrict__ XFT,
                         const unsigned char* __restrict__ BUCK, int m,
                         _Float16* __restrict__ AQre, _Float16* __restrict__ AQim) {
  long idx = (long)blockIdx.x * blockDim.x + threadIdx.x;
  if (idx >= (long)RKE * 128) return;
  int c = (int)(idx & 127);
  long row = idx >> 7;
  int e = (int)(row % Ep);
  long bk = row / Ep;
  int b = (int)(bk / Kn), k = (int)(bk % Kn);
  float xr = 0.f, xi = 0.f;
  if (e < En && BUCK[((long)(b * Cn + c) * Kn + k) * Ep + e] == (unsigned char)m) {
    long src = ((long)(b * Cn + c) * Kn + k) * FSTR;
    xr = XFT[src + e]; xi = XFT[src + 96 + e];
  }
  AQre[idx] = (_Float16)xr;
  AQim[idx] = (_Float16)xi;
}

// ---------------------------------------------------------------------------
// Complex linear y = [A1|A2] @ W^T + b over 256 channels, complex (4 real GEMMs
// folded into one pass; the -im*im term negates the A fragment via negh).
// One wave per 16x16 output tile.
// ---------------------------------------------------------------------------
__global__ void k_clin(const _Float16* __restrict__ A1re, const _Float16* __restrict__ A1im,
                       const _Float16* __restrict__ A2re, const _Float16* __restrict__ A2im,
                       const _Float16* __restrict__ Wtre, const _Float16* __restrict__ Wtim,
                       const float* __restrict__ bre, const float* __restrict__ bim,
                       _Float16* __restrict__ Ore, _Float16* __restrict__ Oim) {
  long mo = (long)blockIdx.x * 16;
  int no = blockIdx.y * 16;
  v8f ar = {}, ai = {};
#pragma unroll
  for (int kt = 0; kt < 8; ++kt) {
    const _Float16* Are = (kt < 4) ? A1re : A2re;
    const _Float16* Aim = (kt < 4) ? A1im : A2im;
    int ka = (kt & 3) * 32;
    v16h fre = load_a(Are + mo * 128 + ka, 128);
    v16h fim = load_a(Aim + mo * 128 + ka, 128);
    v16h wre = load_b(Wtre + (long)kt * 32 * 128 + no, 128);
    v16h wim = load_b(Wtim + (long)kt * 32 * 128 + no, 128);
    ar = wmma16(fre, wre, ar);
    ar = wmma16(negh(fim), wim, ar);   // - im*im
    ai = wmma16(fre, wim, ai);
    ai = wmma16(fim, wre, ai);
  }
  int lane = threadIdx.x & 31, n = lane & 15, mb = 8 * (lane >> 4);
  float br = bre[no + n], bi = bim[no + n];
#pragma unroll
  for (int r = 0; r < 8; ++r) {
    long o = (mo + mb + r) * 128 + no + n;
    Ore[o] = (_Float16)(ar[r] + br);
    Oim[o] = (_Float16)(ai[r] + bi);
  }
}

// ---------------------------------------------------------------------------
// Gate MLP GEMMs (real, 128x128 weights).
// ---------------------------------------------------------------------------
template<bool RELU, bool OUTF32>
__global__ void k_gate_gemm(const _Float16* __restrict__ A, const _Float16* __restrict__ Wt,
                            const float* __restrict__ bias,
                            _Float16* __restrict__ OutH, float* __restrict__ OutF) {
  long mo = (long)blockIdx.x * 16;
  int no = blockIdx.y * 16;
  v8f acc = {};
#pragma unroll
  for (int kt = 0; kt < 4; ++kt) {
    v16h a = load_a(A + mo * 128 + kt * 32, 128);
    v16h b = load_b(Wt + (long)kt * 32 * 128 + no, 128);
    acc = wmma16(a, b, acc);
  }
  int lane = threadIdx.x & 31, n = lane & 15, mb = 8 * (lane >> 4);
  float bb = bias[no + n];
#pragma unroll
  for (int r = 0; r < 8; ++r) {
    float v = acc[r] + bb;
    if (RELU) v = v > 0.f ? v : 0.f;
    long o = (mo + mb + r) * 128 + no + n;
    if (OUTF32) OutF[o] = v; else OutH[o] = (_Float16)v;
  }
}

// Softmax over e (E=85) for each (b,k,c), in place on H2 -> xw.
__global__ void k_gate_softmax(float* __restrict__ H2) {
  int idx = blockIdx.x * blockDim.x + threadIdx.x;
  if (idx >= Bn * Kn * 128) return;
  int c = idx & 127;
  int bk = idx >> 7;
  long base = (long)bk * Ep * 128 + c;
  float mx = -1e30f;
  for (int e = 0; e < En; ++e) mx = fmaxf(mx, H2[base + (long)e * 128]);
  float sum = 0.f;
  for (int e = 0; e < En; ++e) sum += __expf(H2[base + (long)e * 128] - mx);
  float inv = 1.f / sum;
  for (int e = 0; e < En; ++e) {
    long o = base + (long)e * 128;
    H2[o] = __expf(H2[o] - mx) * inv;
  }
}

// ---------------------------------------------------------------------------
// Fourier cross-attention per (b,k).  Whole working set lives in LDS
// (~234 KB, fits CDNA5's 320 KB/WGP).  8 wave32 waves per block; each wave
// owns independent 16x16 WMMA tiles so EXEC is all-ones at every wmma.
// Softmax of |scores| restricted to f<85 (padding masked out).
// ---------------------------------------------------------------------------
__global__ __launch_bounds__(256)
void k_attn(const _Float16* __restrict__ xq_re, const _Float16* __restrict__ xq_im,
            const _Float16* __restrict__ xk_re, const _Float16* __restrict__ xk_im,
            const _Float16* __restrict__ xv_re, const _Float16* __restrict__ xv_im,
            float* __restrict__ cross, int beta) {
  extern __shared__ char smem[];
  _Float16* sq_re = (_Float16*)(smem);
  _Float16* sq_im = (_Float16*)(smem + 24576);
  _Float16* sk_re = (_Float16*)(smem + 49152);
  _Float16* sk_im = (_Float16*)(smem + 73728);
  _Float16* sv_re = (_Float16*)(smem + 98304);
  _Float16* sv_im = (_Float16*)(smem + 122880);
  float*    sc_re = (float*)   (smem + 147456);
  float*    sc_im = (float*)   (smem + 184320);
  _Float16* att   = (_Float16*)(smem + 221184);   // total 239616 B

  int bk = blockIdx.x;
  long base = (long)bk * Ep * 128;
  int tid = threadIdx.x;
  {
    const _Float16* gs[6] = {xq_re + base, xq_im + base, xk_re + base,
                             xk_im + base, xv_re + base, xv_im + base};
    _Float16* gd[6] = {sq_re, sq_im, sk_re, sk_im, sv_re, sv_im};
    for (int a = 0; a < 6; ++a) {
      const uint4* s = (const uint4*)gs[a];
      uint4* d = (uint4*)gd[a];
      for (int i = tid; i < (Ep * 128 * 2) / 16; i += 256) d[i] = s[i];
    }
  }
  __syncthreads();

  int wave = tid >> 5;
  const float scal = 0.10846522890932808f;  // 1/sqrt(85)
  // scores[e,f] = sum_c q[e,c]*k[f,c] (complex, no conj), 36 tiles of 16x16
  for (int t = wave; t < 36; t += 8) {
    int eo = (t / 6) * 16, fo = (t % 6) * 16;
    v8f ar = {}, ai = {};
#pragma unroll
    for (int kt = 0; kt < 4; ++kt) {
      v16h qre = load_a (sq_re + eo * 128 + kt * 32, 128);
      v16h qim = load_a (sq_im + eo * 128 + kt * 32, 128);
      v16h kre = load_bt(sk_re + fo * 128 + kt * 32, 128);
      v16h kim = load_bt(sk_im + fo * 128 + kt * 32, 128);
      ar = wmma16(qre, kre, ar);
      ar = wmma16(negh(qim), kim, ar);   // - im*im
      ai = wmma16(qre, kim, ai);
      ai = wmma16(qim, kre, ai);
    }
    int lane = tid & 31, n = lane & 15, mb = 8 * (lane >> 4);
#pragma unroll
    for (int r = 0; r < 8; ++r) {
      sc_re[(eo + mb + r) * Ep + fo + n] = ar[r] * scal;
      sc_im[(eo + mb + r) * Ep + fo + n] = ai[r] * scal;
    }
  }
  __syncthreads();

  // softmax over |scores| along f<85
  for (int e = tid; e < Ep; e += 256) {
    if (e < En) {
      float mx = -1e30f;
      for (int f = 0; f < En; ++f) {
        float re = sc_re[e * Ep + f], im = sc_im[e * Ep + f];
        mx = fmaxf(mx, sqrtf(re * re + im * im));
      }
      float sum = 0.f;
      for (int f = 0; f < En; ++f) {
        float re = sc_re[e * Ep + f], im = sc_im[e * Ep + f];
        sum += __expf(sqrtf(re * re + im * im) - mx);
      }
      float inv = 1.f / sum;
      for (int f = 0; f < En; ++f) {
        float re = sc_re[e * Ep + f], im = sc_im[e * Ep + f];
        att[e * Ep + f] = (_Float16)(__expf(sqrtf(re * re + im * im) - mx) * inv);
      }
      for (int f = En; f < Ep; ++f) att[e * Ep + f] = (_Float16)0.f;
    } else {
      for (int f = 0; f < Ep; ++f) att[e * Ep + f] = (_Float16)0.f;
    }
  }
  __syncthreads();

  // out = att @ xv (re and im), 48 tiles; accumulate into cross (beta)
  for (int t = wave; t < 48; t += 8) {
    int eo = (t / 8) * 16, co = (t % 8) * 16;
    v8f ore = {}, oim = {};
#pragma unroll
    for (int kt = 0; kt < 3; ++kt) {
      v16h a  = load_a(att + eo * Ep + kt * 32, Ep);
      v16h br = load_b(sv_re + (long)kt * 32 * 128 + co, 128);
      v16h bi = load_b(sv_im + (long)kt * 32 * 128 + co, 128);
      ore = wmma16(a, br, ore);
      oim = wmma16(a, bi, oim);
    }
    int lane = tid & 31, n = lane & 15, mb = 8 * (lane >> 4);
#pragma unroll
    for (int r = 0; r < 8; ++r) {
      long row = (long)bk * Ep + eo + mb + r;
      long o = row * 256 + co + n;
      if (beta) { cross[o] += ore[r]; cross[o + 128] += oim[r]; }
      else      { cross[o]  = ore[r]; cross[o + 128]  = oim[r]; }
    }
  }
}

// ---------------------------------------------------------------------------
// Fused irfft input: irfft(cross*xw) + irfft(resid) = irfft(cross*xw + resid).
// Gathers back to (b,c,k) row layout, f16 for the inverse DFT GEMM.
// ---------------------------------------------------------------------------
__global__ void k_pack_inv(const float* __restrict__ CROSS, const float* __restrict__ XW,
                           const float* __restrict__ XFT,
                           const unsigned char* __restrict__ BUCK,
                           _Float16* __restrict__ AINV) {
  long idx = (long)blockIdx.x * blockDim.x + threadIdx.x;
  if (idx >= (long)RBCK * Ep) return;
  int e = (int)(idx % Ep);
  long row = idx / Ep;                 // (b*Cn + c)*Kn + k
  int k = (int)(row % Kn);
  int c = (int)((row / Kn) % Cn);
  int b = (int)(row / ((long)Cn * Kn));
  float vr = 0.f, vi = 0.f;
  if (e < En) {
    long arow = ((long)(b * Kn + k) * Ep + e);
    float w = XW[arow * 128 + c];
    vr = CROSS[arow * 256 + c] * w;
    vi = CROSS[arow * 256 + 128 + c] * w;
    if (BUCK[row * Ep + e] == 3) {     // residual band
      vr += XFT[row * FSTR + e];
      vi += XFT[row * FSTR + 96 + e];
    }
  }
  AINV[row * FSTR + e]      = (_Float16)vr;
  AINV[row * FSTR + 96 + e] = (_Float16)vi;
}

// ---------------------------------------------------------------------------
// Host orchestration.  Workspace demand ~1.54 GB (trivial vs 432 GiB HBM).
// ---------------------------------------------------------------------------
extern "C" void kernel_launch(void* const* d_in, const int* in_sizes, int n_in,
                              void* d_out, int out_size, void* d_ws, size_t ws_size,
                              hipStream_t stream) {
  (void)in_sizes; (void)n_in; (void)out_size; (void)ws_size;
  const float* x_d   = (const float*)d_in[0];
  const float* bs_c  = (const float*)d_in[1];
  const float* Wq_re = (const float*)d_in[2],  *Wq_im = (const float*)d_in[3];
  const float* bq_re = (const float*)d_in[4],  *bq_im = (const float*)d_in[5];
  const float* Wk_re = (const float*)d_in[6],  *Wk_im = (const float*)d_in[7];
  const float* bk_re = (const float*)d_in[8],  *bk_im = (const float*)d_in[9];
  const float* Wv_re = (const float*)d_in[10], *Wv_im = (const float*)d_in[11];
  const float* bv_re = (const float*)d_in[12], *bv_im = (const float*)d_in[13];
  const float* fw1 = (const float*)d_in[14], *fb1 = (const float*)d_in[15];
  const float* fw2 = (const float*)d_in[16], *fb2 = (const float*)d_in[17];
  float* out = (float*)d_out;

  char* ws = (char*)d_ws;
  size_t off = 0;
  auto take = [&](size_t bytes) -> void* {
    void* p = ws + off;
    off += (bytes + 255) & ~(size_t)255;
    return p;
  };
  float*    XFT  = (float*)take((size_t)RBCK * FSTR * 4);
  float*    BSF  = (float*)take((size_t)RBCK * FSTR * 4);
  unsigned char* BUCK = (unsigned char*)take((size_t)RBCK * Ep);
  _Float16* AXre = (_Float16*)take((size_t)RKE * 128 * 2);
  _Float16* AXim = (_Float16*)take((size_t)RKE * 128 * 2);
  _Float16* ABre = (_Float16*)take((size_t)RKE * 128 * 2);
  _Float16* ABim = (_Float16*)take((size_t)RKE * 128 * 2);
  _Float16* AG   = (_Float16*)take((size_t)RKE * 128 * 2);
  _Float16* AQre = (_Float16*)take((size_t)RKE * 128 * 2);
  _Float16* AQim = (_Float16*)take((size_t)RKE * 128 * 2);
  _Float16* XKre = (_Float16*)take((size_t)RKE * 128 * 2);
  _Float16* XKim = (_Float16*)take((size_t)RKE * 128 * 2);
  _Float16* XVre = (_Float16*)take((size_t)RKE * 128 * 2);
  _Float16* XVim = (_Float16*)take((size_t)RKE * 128 * 2);
  _Float16* XQre = (_Float16*)take((size_t)RKE * 128 * 2);
  _Float16* XQim = (_Float16*)take((size_t)RKE * 128 * 2);
  _Float16* H1   = (_Float16*)take((size_t)RKE * 128 * 2);
  float*    H2   = (float*)take((size_t)RKE * 128 * 4);
  float*    CROSS= (float*)take((size_t)RKE * 256 * 4);
  _Float16* AINV = (_Float16*)take((size_t)RBCK * FSTR * 2);
  _Float16* TFWD = (_Float16*)take(192 * 192 * 2);
  _Float16* TINV = (_Float16*)take(192 * 176 * 2);
  _Float16* WQtre = (_Float16*)take(256 * 128 * 2);
  _Float16* WQtim = (_Float16*)take(256 * 128 * 2);
  _Float16* WKtre = (_Float16*)take(256 * 128 * 2);
  _Float16* WKtim = (_Float16*)take(256 * 128 * 2);
  _Float16* WVtre = (_Float16*)take(256 * 128 * 2);
  _Float16* WVtim = (_Float16*)take(256 * 128 * 2);
  _Float16* FW1t  = (_Float16*)take(128 * 128 * 2);
  _Float16* FW2t  = (_Float16*)take(128 * 128 * 2);

  dim3 b32(32), b256(256);

  k_prep<<<dim3(512), b256, 0, stream>>>(TFWD, TINV, Wq_re, Wq_im, Wk_re, Wk_im,
                                         Wv_re, Wv_im, fw1, fw2,
                                         WQtre, WQtim, WKtre, WKtim, WVtre, WVtim,
                                         FW1t, FW2t);

  // rfft via WMMA GEMM against f16 twiddles (N=192 packs re|im)
  k_dft_gemm<float><<<dim3(RBCK / 16, 12), b32, 0, stream>>>(
      x_d, Ln, Ln, 6, TFWD, 192, XFT, FSTR, 192);
  k_dft_gemm<float><<<dim3(RBCK / 16, 12), b32, 0, stream>>>(
      bs_c, Ln, Ln, 6, TFWD, 192, BSF, FSTR, 192);

  k_topk<<<dim3(RBCK / 256), b256, 0, stream>>>(XFT, BUCK);

  unsigned packBlocks = (unsigned)(((long)RKE * 128) / 256);
  k_pack_kv<<<dim3(packBlocks), b256, 0, stream>>>(XFT, BSF, AXre, AXim, ABre, ABim, AG);

  // xk, xv once (reference recomputes them per sub; they are identical)
  k_clin<<<dim3(RKE / 16, 8), b32, 0, stream>>>(AXre, AXim, ABre, ABim,
                                                WKtre, WKtim, bk_re, bk_im, XKre, XKim);
  k_clin<<<dim3(RKE / 16, 8), b32, 0, stream>>>(AXre, AXim, ABre, ABim,
                                                WVtre, WVtim, bv_re, bv_im, XVre, XVim);

  // frequency-gating MLP + softmax over E
  k_gate_gemm<true,  false><<<dim3(RKE / 16, 8), b32, 0, stream>>>(AG, FW1t, fb1, H1, (float*)nullptr);
  k_gate_gemm<false, true ><<<dim3(RKE / 16, 8), b32, 0, stream>>>(H1, FW2t, fb2, (_Float16*)nullptr, H2);
  k_gate_softmax<<<dim3((Bn * Kn * 128) / 256), b256, 0, stream>>>(H2);

  // per-sub-band Q projection + attention, accumulated into CROSS
  for (int m = 0; m < 3; ++m) {
    k_pack_q<<<dim3(packBlocks), b256, 0, stream>>>(XFT, BUCK, m, AQre, AQim);
    k_clin<<<dim3(RKE / 16, 8), b32, 0, stream>>>(AQre, AQim, ABre, ABim,
                                                  WQtre, WQtim, bq_re, bq_im, XQre, XQim);
    k_attn<<<dim3(Bn * Kn), b256, 239616, stream>>>(XQre, XQim, XKre, XKim,
                                                    XVre, XVim, CROSS, m > 0 ? 1 : 0);
  }

  // fused irfft(cross*xw + resid) via one inverse-DFT GEMM straight into d_out
  k_pack_inv<<<dim3((unsigned)(((long)RBCK * Ep) / 256)), b256, 0, stream>>>(
      CROSS, H2, XFT, BUCK, AINV);
  k_dft_gemm<_Float16><<<dim3(RBCK / 16, 11), b32, 0, stream>>>(
      AINV, FSTR, FSTR, 6, TINV, 176, out, Ln, Ln);
}
